// MambaLayer_22393959481420
// MI455X (gfx1250) — compile-verified, matrix operand order checked by reading, not run on hardware
//
#include <hip/hip_runtime.h>
#include <hip/hip_bf16.h>
#include <math.h>

#define B_     4
#define NQ_    4096
#define NI_    8192
#define D_     256
#define K_     8
#define NL_    2
#define INTER_ 512
#define NH_    8
#define HP_    64
#define NS_    64
#define CONV_  640     // INTER + 2*NS
#define PROJ_  1160    // INTER + CONV + NH
#define HFF_   1024
#define KER_   4

typedef __attribute__((ext_vector_type(16))) _Float16 v16h;
typedef __attribute__((ext_vector_type(8)))  float    v8f;

#if __has_builtin(__builtin_amdgcn_s_wait_asynccnt)
#define WAIT_ASYNC(n) __builtin_amdgcn_s_wait_asynccnt(n)
#else
#define WAIT_ASYNC(n) asm volatile("s_wait_asynccnt %0" :: "i"(n) : "memory")
#endif

// ---------------------------------------------------------------------------
// fp32 -> fp16 conversion (weights once per call, activations at producers)
// ---------------------------------------------------------------------------
__global__ __launch_bounds__(256)
void cvt16_kernel(const float* __restrict__ in, _Float16* __restrict__ out,
                  long long n)
{
    const long long i = ((long long)blockIdx.x * 256 + threadIdx.x) * 8;
    if (i + 8 <= n) {
        const float4 f0 = *(const float4*)(in + i);
        const float4 f1 = *(const float4*)(in + i + 4);
        union { _Float16 h[8]; uint4 u; } p;
        p.h[0]=(_Float16)f0.x; p.h[1]=(_Float16)f0.y;
        p.h[2]=(_Float16)f0.z; p.h[3]=(_Float16)f0.w;
        p.h[4]=(_Float16)f1.x; p.h[5]=(_Float16)f1.y;
        p.h[6]=(_Float16)f1.z; p.h[7]=(_Float16)f1.w;
        *(uint4*)(out + i) = p.u;
    } else {
        for (long long j = i; j < n; ++j) out[j] = (_Float16)in[j];
    }
}

// ---------------------------------------------------------------------------
// WMMA GEMM, f16 operands in memory, async global->LDS double-buffered tiles.
//   C[M,N] = act( gather(A)[M,K] * B[N,K]^T + bias ) + Res
// A row-major f16 (lda), B row-major N x K f16 (ld=Kd), C row-major fp32
// and/or f16. Block 256 thr = 8 waves; tile 64x64; wave tile 16x32.
// ---------------------------------------------------------------------------
__global__ __launch_bounds__(256)
void gemm_f16wmma(const _Float16* __restrict__ A, int lda, long long strideA,
                  const int* __restrict__ aidx, long long strideIdx,
                  const _Float16* __restrict__ Bw,
                  float* __restrict__ C, long long strideC,
                  _Float16* __restrict__ C16, long long strideC16,
                  int ldc,
                  const float* __restrict__ bias,
                  const float* __restrict__ Res, long long strideRes,
                  int M, int N, int Kd, int act)
{
    // 80-byte row stride: 16B aligned for b128 async writes, conflict padding
    __shared__ __align__(16) _Float16 As[2][64][40];
    __shared__ __align__(16) _Float16 Bs[2][64][40];

    const int tid = threadIdx.x;
    const int z   = blockIdx.z;
    const int mb  = blockIdx.y * 64;
    const int nb  = blockIdx.x * 64;

    const _Float16* Ab = A + (long long)z * strideA;
    float*        Cb  = C   ? C   + (long long)z * strideC   : nullptr;
    _Float16*     C16b= C16 ? C16 + (long long)z * strideC16 : nullptr;
    const float*  Rb  = Res ? Res + (long long)z * strideRes : nullptr;

    // staging: 4 threads per tile row, one 16-byte async chunk each
    const int srow   = tid >> 2;
    const int schunk = (tid & 3) << 3;          // halves: 0,8,16,24
    int gm = mb + srow; if (gm > M - 1) gm = M - 1;   // clamp, never store OOB
    const long long arow =
        aidx ? (long long)aidx[(long long)z * strideIdx + gm] : (long long)gm;
    int gn = nb + srow; if (gn > N - 1) gn = N - 1;
    const _Float16* aptr = Ab + arow * (long long)lda + schunk;
    const _Float16* bptr = Bw + (long long)gn * Kd + schunk;

    const int wave = tid >> 5, lane = tid & 31;
    const int wm = (wave >> 1) << 4;   // 0,16,32,48
    const int wn = (wave & 1) << 5;    // 0,32
    const int r  = lane & 15, g = lane >> 4;

    v8f acc0 = {0.f,0.f,0.f,0.f,0.f,0.f,0.f,0.f};
    v8f acc1 = acc0;

    auto stage = [&](int kc, int buf) {
        const unsigned la = (unsigned)(size_t)&As[buf][srow][schunk];
        const unsigned lb = (unsigned)(size_t)&Bs[buf][srow][schunk];
        asm volatile("global_load_async_to_lds_b128 %0, %1, off"
                     :: "v"(la), "v"((unsigned long long)(size_t)(aptr + kc))
                     : "memory");
        asm volatile("global_load_async_to_lds_b128 %0, %1, off"
                     :: "v"(lb), "v"((unsigned long long)(size_t)(bptr + kc))
                     : "memory");
    };

    const int nk = Kd >> 5;
    stage(0, 0);
    for (int i = 0; i < nk; ++i) {
        const int buf = i & 1;
        if (i + 1 < nk) {
            stage((i + 1) << 5, buf ^ 1);
            WAIT_ASYNC(2);          // retire stage i, keep i+1 in flight
        } else {
            WAIT_ASYNC(0);
        }
        __syncthreads();            // stage-i tiles visible to all waves

        union { v16h v; unsigned u[8]; } fa, fb0, fb1;
        #pragma unroll
        for (int vv = 0; vv < 8; ++vv) {
            const int k0 = ((vv & 3) << 1) + ((vv >> 2) << 4) + (g << 3);
            fa.u [vv] = *(const unsigned*)&As[buf][wm + r][k0];
            fb0.u[vv] = *(const unsigned*)&Bs[buf][wn + r][k0];
            fb1.u[vv] = *(const unsigned*)&Bs[buf][wn + 16 + r][k0];
        }
        acc0 = __builtin_amdgcn_wmma_f32_16x16x32_f16(
                   false, fa.v, false, fb0.v, (short)0, acc0, false, false);
        acc1 = __builtin_amdgcn_wmma_f32_16x16x32_f16(
                   false, fa.v, false, fb1.v, (short)0, acc1, false, false);
        __syncthreads();            // reads done before buf is re-staged
    }

    // ---- epilogue ---------------------------------------------------------
    #pragma unroll
    for (int rr = 0; rr < 8; ++rr) {
        const int gmr = mb + wm + rr + (g << 3);
        if (gmr >= M) continue;
        #pragma unroll
        for (int half = 0; half < 2; ++half) {
            const int gnc = nb + wn + (half << 4) + r;
            if (gnc >= N) continue;
            float val = half ? acc1[rr] : acc0[rr];
            if (bias) val += bias[gnc];
            if (act == 1)
                val = 0.5f * val * (1.f + erff(val * 0.70710678118f));
            if (Rb) val += Rb[(long long)gmr * ldc + gnc];
            if (Cb)   Cb  [(long long)gmr * ldc + gnc] = val;
            if (C16b) C16b[(long long)gmr * ldc + gnc] = (_Float16)val;
        }
    }
}

// ---------------------------------------------------------------------------
// KNN: per query keep 8 nearest of 8192 points; writes GLOBAL row indices
// (b*NI + i) so the v-GEMM can gather inst_feats directly.
// ---------------------------------------------------------------------------
__global__ __launch_bounds__(256)
void knn_kernel(const float* __restrict__ qpos, const float* __restrict__ spc,
                int* __restrict__ idxg)
{
    const int b = blockIdx.y;
    const int q = blockIdx.x * 256 + threadIdx.x;
    const float* qp = qpos + ((long long)b * NQ_ + q) * 3;
    const float qx = qp[0], qy = qp[1], qz = qp[2];

    float bd[K_]; int bi[K_];
    #pragma unroll
    for (int k = 0; k < K_; ++k) { bd[k] = 1e30f; bi[k] = 0; }

    __shared__ float sp[512 * 3];
    for (int tile = 0; tile < NI_; tile += 512) {
        for (int i = threadIdx.x; i < 512 * 3; i += 256)
            sp[i] = spc[(long long)b * NI_ * 3 + (long long)tile * 3 + i];
        __syncthreads();
        for (int j = 0; j < 512; ++j) {
            const float dx = qx - sp[j*3+0];
            const float dy = qy - sp[j*3+1];
            const float dz = qz - sp[j*3+2];
            const float d  = dx*dx + dy*dy + dz*dz;
            if (d < bd[K_-1]) {
                bd[K_-1] = d; bi[K_-1] = tile + j;
                #pragma unroll
                for (int t = K_-1; t > 0; --t) {
                    if (bd[t] < bd[t-1]) {
                        float td = bd[t]; bd[t] = bd[t-1]; bd[t-1] = td;
                        int   ti = bi[t]; bi[t] = bi[t-1]; bi[t-1] = ti;
                    }
                }
            }
        }
        __syncthreads();
    }
    #pragma unroll
    for (int k = 0; k < K_; ++k)
        idxg[((long long)b * NQ_ + q) * K_ + k] = b * NI_ + bi[k];
}

// ---------------------------------------------------------------------------
// Attention combine: kk = softmax(q@w_k^T + w_b); w = q * sum_k kk*v  (f16 out)
// ---------------------------------------------------------------------------
__global__ __launch_bounds__(256)
void attn_kernel(const float* __restrict__ qb, const float* __restrict__ vb,
                 const float* __restrict__ w_k, const float* __restrict__ w_b,
                 _Float16* __restrict__ wout)
{
    const int q = blockIdx.x, b = blockIdx.y, t = threadIdx.x;
    const long long row = (long long)b * NQ_ + q;
    __shared__ float slog[K_];
    __shared__ float skk[K_];
    if (t < K_) slog[t] = w_b[t];
    __syncthreads();
    const float qd = qb[row * D_ + t];
    #pragma unroll
    for (int k = 0; k < K_; ++k)
        atomicAdd(&slog[k], qd * w_k[k * D_ + t]);
    __syncthreads();
    if (t == 0) {
        float mx = slog[0];
        #pragma unroll
        for (int k = 1; k < K_; ++k) mx = fmaxf(mx, slog[k]);
        float sum = 0.f;
        #pragma unroll
        for (int k = 0; k < K_; ++k) { skk[k] = __expf(slog[k] - mx); sum += skk[k]; }
        const float inv = 1.f / sum;
        #pragma unroll
        for (int k = 0; k < K_; ++k) skk[k] *= inv;
    }
    __syncthreads();
    float acc = 0.f;
    #pragma unroll
    for (int k = 0; k < K_; ++k)
        acc += skk[k] * vb[(row * K_ + k) * (long long)D_ + t];
    wout[row * D_ + t] = (_Float16)(qd * acc);
}

// ---------------------------------------------------------------------------
// LayerNorm over D=256, optional residual; fp32 and/or f16 outputs.
// ---------------------------------------------------------------------------
__global__ __launch_bounds__(256)
void ln_kernel(const float* __restrict__ a, const float* __restrict__ res,
               const float* __restrict__ gg, const float* __restrict__ bb,
               float* __restrict__ out, _Float16* __restrict__ out16)
{
    const long long row = blockIdx.x;
    const int d = threadIdx.x;
    float v = a[row * D_ + d];
    if (res) v += res[row * D_ + d];
    __shared__ float s1[256], s2[256];
    s1[d] = v; s2[d] = v * v;
    __syncthreads();
    for (int s = 128; s > 0; s >>= 1) {
        if (d < s) { s1[d] += s1[d + s]; s2[d] += s2[d + s]; }
        __syncthreads();
    }
    const float m   = s1[0] * (1.f / D_);
    const float var = s2[0] * (1.f / D_) - m * m;
    const float rv  = (v - m) * rsqrtf(var + 1e-5f) * gg[d] + bb[d];
    if (out)   out  [row * D_ + d] = rv;
    if (out16) out16[row * D_ + d] = (_Float16)rv;
}

// ---------------------------------------------------------------------------
// Morton pipeline
// ---------------------------------------------------------------------------
__global__ __launch_bounds__(256)
void minmax_kernel(const float* __restrict__ qpos,
                   float* __restrict__ mn, float* __restrict__ span)
{
    const int b = blockIdx.x, t = threadIdx.x;
    float lmn[3] = {1e30f, 1e30f, 1e30f};
    float lmx[3] = {-1e30f, -1e30f, -1e30f};
    const float* base = qpos + (long long)b * NQ_ * 3;
    for (int i = t; i < NQ_; i += 256) {
        #pragma unroll
        for (int d = 0; d < 3; ++d) {
            const float v = base[i * 3 + d];
            lmn[d] = fminf(lmn[d], v);
            lmx[d] = fmaxf(lmx[d], v);
        }
    }
    __shared__ float smn[3][256], smx[3][256];
    #pragma unroll
    for (int d = 0; d < 3; ++d) { smn[d][t] = lmn[d]; smx[d][t] = lmx[d]; }
    __syncthreads();
    for (int s = 128; s > 0; s >>= 1) {
        if (t < s) {
            #pragma unroll
            for (int d = 0; d < 3; ++d) {
                smn[d][t] = fminf(smn[d][t], smn[d][t + s]);
                smx[d][t] = fmaxf(smx[d][t], smx[d][t + s]);
            }
        }
        __syncthreads();
    }
    if (t < 3) {
        mn[b * 3 + t]   = smn[t][0];
        span[b * 3 + t] = smx[t][0] - smn[t][0] + 1e-6f;
    }
}

__device__ inline unsigned part1by2(unsigned x)
{
    x &= 1023u;
    x = (x | (x << 16)) & 0xFF0000FFu;
    x = (x | (x << 8))  & 0x0F00F00Fu;
    x = (x | (x << 4))  & 0xC30C30C3u;
    x = (x | (x << 2))  & 0x49249249u;
    return x;
}

__global__ __launch_bounds__(256)
void morton_kernel(const float* __restrict__ qpos, const float* __restrict__ mn,
                   const float* __restrict__ span,
                   unsigned long long* __restrict__ keys)
{
    const int b = blockIdx.y;
    const int q = blockIdx.x * 256 + threadIdx.x;
    const float* p = qpos + ((long long)b * NQ_ + q) * 3;
    unsigned gi[3];
    #pragma unroll
    for (int d = 0; d < 3; ++d) {
        float v = (p[d] - mn[b * 3 + d]) / span[b * 3 + d] * 1023.f;
        v = fminf(fmaxf(v, 0.f), 1023.f);
        gi[d] = (unsigned)(int)v;
    }
    const unsigned c1 = part1by2(gi[0]) | (part1by2(gi[1]) << 1) | (part1by2(gi[2]) << 2);
    const unsigned c2 = part1by2(gi[2]) | (part1by2(gi[1]) << 1) | (part1by2(gi[0]) << 2);
    keys[((long long)(b * 2 + 0)) * NQ_ + q] = ((unsigned long long)c1 << 32) | (unsigned)q;
    keys[((long long)(b * 2 + 1)) * NQ_ + q] = ((unsigned long long)c2 << 32) | (unsigned)q;
}

__global__ __launch_bounds__(1024)
void sort_kernel(const unsigned long long* __restrict__ keys,
                 int* __restrict__ ordg, int* __restrict__ inv)
{
    const int blk = blockIdx.x;          // b*2 + dir
    const int t = threadIdx.x;
    __shared__ unsigned long long sk[NQ_];
    for (int i = t; i < NQ_; i += 1024)
        sk[i] = keys[(long long)blk * NQ_ + i];
    __syncthreads();
    for (int k = 2; k <= NQ_; k <<= 1) {
        for (int j = k >> 1; j > 0; j >>= 1) {
            for (int i = t; i < NQ_; i += 1024) {
                const int l = i ^ j;
                if (l > i) {
                    const bool up = ((i & k) == 0);
                    const unsigned long long a = sk[i], c = sk[l];
                    if (up ? (a > c) : (a < c)) { sk[i] = c; sk[l] = a; }
                }
            }
            __syncthreads();
        }
    }
    const int b = blk >> 1;
    for (int i = t; i < NQ_; i += 1024) {
        const int v = (int)(sk[i] & 0xFFFFFFFFull);
        ordg[(long long)blk * NQ_ + i] = b * NQ_ + v;   // global row into xn16
        inv[(long long)blk * NQ_ + v]  = i;             // local inverse perm
    }
}

// ---------------------------------------------------------------------------
// Causal depthwise conv (KER=4) + SiLU over CONV channels of zxbcdt.
// ---------------------------------------------------------------------------
__global__ __launch_bounds__(256)
void conv_kernel(const float* __restrict__ zx, const float* __restrict__ cw,
                 const float* __restrict__ cb, float* __restrict__ xbc)
{
    const int z = blockIdx.y;
    const long long e = (long long)blockIdx.x * 256 + threadIdx.x;
    if (e >= (long long)NQ_ * CONV_) return;
    const int s = (int)(e / CONV_);
    const int c = (int)(e % CONV_);
    const float* zb = zx + (long long)z * NQ_ * PROJ_;
    float acc = cb[c];
    #pragma unroll
    for (int j = 0; j < KER_; ++j) {
        const int sp = s - (KER_ - 1) + j;
        if (sp >= 0)
            acc += zb[(long long)sp * PROJ_ + INTER_ + c] * cw[c * KER_ + j];
    }
    xbc[((long long)z * NQ_ + s) * CONV_ + c] = acc / (1.f + __expf(-acc));
}

__global__ __launch_bounds__(256)
void dt_kernel(const float* __restrict__ zx, const float* __restrict__ dt_bias,
               float* __restrict__ dtb)
{
    const int z = blockIdx.y;
    const int e = blockIdx.x * 256 + threadIdx.x;   // s*NH + h
    const int s = e >> 3, h = e & 7;
    const float x = zx[((long long)z * NQ_ + s) * PROJ_ + INTER_ + CONV_ + h] + dt_bias[h];
    dtb[((long long)z * NQ_ + s) * NH_ + h] = (x > 20.f) ? x : log1pf(__expf(x));
}

// ---------------------------------------------------------------------------
// Selective-scan: one block per (head, z=(b,p)); state (64x64) in registers.
// ---------------------------------------------------------------------------
__global__ __launch_bounds__(256)
void scan_kernel(const float* __restrict__ xbc, const float* __restrict__ dtb,
                 const float* __restrict__ A_log, const float* __restrict__ Dp,
                 float* __restrict__ ys)
{
    const int h = blockIdx.x, z = blockIdx.y, t = threadIdx.x;
    const int hp = t & 63, nbk = t >> 6;
    const float Aa = -__expf(A_log[h]);
    const float Dv = Dp[h];
    float st[16];
    #pragma unroll
    for (int i = 0; i < 16; ++i) st[i] = 0.f;

    __shared__ float sx[64], sB[64], sC[64], sp[256];
    __shared__ float sdt;
    const long long base = (long long)z * NQ_;

    for (int s = 0; s < NQ_; ++s) {
        const long long rowc = (base + s) * CONV_;
        if (t < 64)        sx[t]       = xbc[rowc + h * HP_ + t];
        else if (t < 128)  sB[t - 64]  = xbc[rowc + INTER_ + (t - 64)];
        else if (t < 192)  sC[t - 128] = xbc[rowc + INTER_ + NS_ + (t - 128)];
        else if (t == 192) sdt         = dtb[(base + s) * NH_ + h];
        __syncthreads();

        const float dtv = sdt;
        const float dA  = __expf(dtv * Aa);
        const float xv  = sx[hp];
        const float dx  = dtv * xv;
        float part = 0.f;
        #pragma unroll
        for (int i = 0; i < 16; ++i) {
            const int ns = nbk * 16 + i;
            st[i] = st[i] * dA + dx * sB[ns];
            part += st[i] * sC[ns];
        }
        sp[t] = part;
        __syncthreads();
        if (t < 64) {
            const float y = sp[t] + sp[t + 64] + sp[t + 128] + sp[t + 192]
                          + Dv * sx[t];
            ys[(base + s) * INTER_ + h * HP_ + t] = y;
        }
        __syncthreads();
    }
}

// ---------------------------------------------------------------------------
// h = RMSNorm(y * silu(gate)) * norm_w  -> f16 (feeds out-proj GEMM)
// ---------------------------------------------------------------------------
__global__ __launch_bounds__(256)
void gated_rms_kernel(const float* __restrict__ ys, const float* __restrict__ zx,
                      const float* __restrict__ nw, _Float16* __restrict__ hout)
{
    const int s = blockIdx.x, z = blockIdx.y, t = threadIdx.x;
    const long long row = (long long)z * NQ_ + s;
    float g0 = zx[row * PROJ_ + t];
    float g1 = zx[row * PROJ_ + t + 256];
    const float h0 = ys[row * INTER_ + t]       * g0 / (1.f + __expf(-g0));
    const float h1 = ys[row * INTER_ + t + 256] * g1 / (1.f + __expf(-g1));
    __shared__ float sr[256];
    sr[t] = h0 * h0 + h1 * h1;
    __syncthreads();
    for (int sdx = 128; sdx > 0; sdx >>= 1) {
        if (t < sdx) sr[t] += sr[t + sdx];
        __syncthreads();
    }
    const float sc = rsqrtf(sr[0] * (1.f / INTER_) + 1e-5f);
    hout[row * INTER_ + t]       = (_Float16)(h0 * sc * nw[t]);
    hout[row * INTER_ + t + 256] = (_Float16)(h1 * sc * nw[t + 256]);
}

// ---------------------------------------------------------------------------
// x = LN( x + 0.5*(ydir[p=0][inv0[s]] + ydir[p=1][inv1[s]]) )  (in place)
// ---------------------------------------------------------------------------
__global__ __launch_bounds__(256)
void combine_ln_kernel(float* __restrict__ x, _Float16* __restrict__ x16,
                       const float* __restrict__ ydir,
                       const int* __restrict__ inv,
                       const float* __restrict__ gg, const float* __restrict__ bb)
{
    const int s = blockIdx.x, b = blockIdx.y, d = threadIdx.x;
    const long long row = (long long)b * NQ_ + s;
    const int i0 = inv[(long long)(b * 2 + 0) * NQ_ + s];
    const int i1 = inv[(long long)(b * 2 + 1) * NQ_ + s];
    const float back =
        0.5f * ( ydir[((long long)(b * 2 + 0) * NQ_ + i0) * D_ + d]
               + ydir[((long long)(b * 2 + 1) * NQ_ + i1) * D_ + d] );
    const float v = x[row * D_ + d] + back;
    __shared__ float s1[256], s2[256];
    s1[d] = v; s2[d] = v * v;
    __syncthreads();
    for (int st = 128; st > 0; st >>= 1) {
        if (d < st) { s1[d] += s1[d + st]; s2[d] += s2[d + st]; }
        __syncthreads();
    }
    const float m   = s1[0] * (1.f / D_);
    const float var = s2[0] * (1.f / D_) - m * m;
    const float rv  = (v - m) * rsqrtf(var + 1e-5f) * gg[d] + bb[d];
    x  [row * D_ + d] = rv;
    x16[row * D_ + d] = (_Float16)rv;
}

// ---------------------------------------------------------------------------
extern "C" void kernel_launch(void* const* d_in, const int* in_sizes, int n_in,
                              void* d_out, int out_size, void* d_ws, size_t ws_size,
                              hipStream_t stream)
{
    (void)in_sizes; (void)n_in; (void)out_size; (void)ws_size;
    const float* queries  = (const float*)d_in[0];
    const float* qpos     = (const float*)d_in[1];
    const float* inst     = (const float*)d_in[2];
    const float* spc      = (const float*)d_in[3];
    const float* w_q      = (const float*)d_in[4];
    const float* w_v      = (const float*)d_in[5];
    const float* w_o      = (const float*)d_in[6];
    const float* w_k      = (const float*)d_in[7];
    const float* w_b      = (const float*)d_in[8];
    const float* agg_g    = (const float*)d_in[9];
    const float* agg_b    = (const float*)d_in[10];
    const float* pre_g    = (const float*)d_in[11];
    const float* pre_b    = (const float*)d_in[12];
    const float* fin_g    = (const float*)d_in[13];
    const float* fin_b    = (const float*)d_in[14];
    const float* m_in_w   = (const float*)d_in[15];
    const float* m_conv_w = (const float*)d_in[16];
    const float* m_conv_b = (const float*)d_in[17];
    const float* m_dt_b   = (const float*)d_in[18];
    const float* m_A_log  = (const float*)d_in[19];
    const float* m_Dp     = (const float*)d_in[20];
    const float* m_norm_w = (const float*)d_in[21];
    const float* m_out_w  = (const float*)d_in[22];
    const float* f_w1     = (const float*)d_in[23];
    const float* f_b1     = (const float*)d_in[24];
    const float* f_w2     = (const float*)d_in[25];
    const float* f_b2     = (const float*)d_in[26];
    const float* fl_g     = (const float*)d_in[27];
    const float* fl_b     = (const float*)d_in[28];
    float* out = (float*)d_out;

    size_t off = 0;
    auto alloc = [&](size_t bytes) -> void* {
        void* p = (char*)d_ws + off;
        off += (bytes + 255) & ~(size_t)255;
        return p;
    };
    // fp32 work buffers
    int*   idxg  = (int*)  alloc((size_t)B_*NQ_*K_*4);
    float* qb    = (float*)alloc((size_t)B_*NQ_*D_*4);
    float* vb    = (float*)alloc((size_t)B_*NQ_*K_*D_*4);
    float* xb    = (float*)alloc((size_t)B_*NQ_*D_*4);
    float* xn    = (float*)alloc((size_t)B_*NQ_*D_*4);
    float* mnb   = (float*)alloc((size_t)B_*3*4);
    float* spanb = (float*)alloc((size_t)B_*3*4);
    unsigned long long* keys = (unsigned long long*)alloc((size_t)B_*2*NQ_*8);
    int*   ordg  = (int*)  alloc((size_t)B_*2*NQ_*4);
    int*   invb  = (int*)  alloc((size_t)B_*2*NQ_*4);
    float* zx    = (float*)alloc((size_t)B_*2*NQ_*PROJ_*4);
    float* xbc   = (float*)alloc((size_t)B_*2*NQ_*CONV_*4);
    float* dtbuf = (float*)alloc((size_t)B_*2*NQ_*NH_*4);
    float* ysb   = (float*)alloc((size_t)B_*2*NQ_*INTER_*4);
    float* ydir  = (float*)alloc((size_t)B_*2*NQ_*D_*4);
    float* ffo   = (float*)alloc((size_t)B_*NQ_*D_*4);
    // f16 operand buffers
    _Float16* queries16 = (_Float16*)alloc((size_t)B_*NQ_*D_*2);
    _Float16* inst16    = (_Float16*)alloc((size_t)B_*NI_*D_*2);
    _Float16* w_q16     = (_Float16*)alloc((size_t)D_*D_*2);
    _Float16* w_v16     = (_Float16*)alloc((size_t)D_*D_*2);
    _Float16* w_o16     = (_Float16*)alloc((size_t)D_*D_*2);
    _Float16* m_in_w16  = (_Float16*)alloc((size_t)NL_*PROJ_*D_*2);
    _Float16* m_out_w16 = (_Float16*)alloc((size_t)NL_*D_*INTER_*2);
    _Float16* f_w116    = (_Float16*)alloc((size_t)HFF_*D_*2);
    _Float16* f_w216    = (_Float16*)alloc((size_t)D_*HFF_*2);
    _Float16* wbuf16    = (_Float16*)alloc((size_t)B_*NQ_*D_*2);
    _Float16* xn16      = (_Float16*)alloc((size_t)B_*NQ_*D_*2);
    _Float16* xb16      = (_Float16*)alloc((size_t)B_*NQ_*D_*2);
    _Float16* hbuf16    = (_Float16*)alloc((size_t)B_*2*NQ_*INTER_*2);
    _Float16* ffh16     = (_Float16*)alloc((size_t)B_*NQ_*HFF_*2);

    const dim3 blk256(256);
    auto cvt = [&](const float* src, _Float16* dst, long long n) {
        cvt16_kernel<<<dim3((unsigned)((n/8 + 255)/256)), blk256, 0, stream>>>(src, dst, n);
    };

    // --- operand conversion ----------------------------------------------
    cvt(queries, queries16, (long long)B_*NQ_*D_);
    cvt(inst,    inst16,    (long long)B_*NI_*D_);
    cvt(w_q,     w_q16,     (long long)D_*D_);
    cvt(w_v,     w_v16,     (long long)D_*D_);
    cvt(w_o,     w_o16,     (long long)D_*D_);
    cvt(m_in_w,  m_in_w16,  (long long)NL_*PROJ_*D_);
    cvt(m_out_w, m_out_w16, (long long)NL_*D_*INTER_);
    cvt(f_w1,    f_w116,    (long long)HFF_*D_);
    cvt(f_w2,    f_w216,    (long long)D_*HFF_);

    // --- KNN + attention aggregation --------------------------------------
    knn_kernel<<<dim3(NQ_/256, B_), blk256, 0, stream>>>(qpos, spc, idxg);

    gemm_f16wmma<<<dim3(D_/64, NQ_/64, B_), blk256, 0, stream>>>(
        queries16, D_, (long long)NQ_*D_, nullptr, 0,
        w_q16, qb, (long long)NQ_*D_, nullptr, 0, D_,
        nullptr, nullptr, 0, NQ_, D_, D_, 0);

    gemm_f16wmma<<<dim3(D_/64, (NQ_*K_)/64, B_), blk256, 0, stream>>>(
        inst16, D_, 0, idxg, (long long)NQ_*K_,
        w_v16, vb, (long long)NQ_*K_*D_, nullptr, 0, D_,
        nullptr, nullptr, 0, NQ_*K_, D_, D_, 0);

    attn_kernel<<<dim3(NQ_, B_), blk256, 0, stream>>>(qb, vb, w_k, w_b, wbuf16);

    gemm_f16wmma<<<dim3(D_/64, NQ_/64, B_), blk256, 0, stream>>>(
        wbuf16, D_, (long long)NQ_*D_, nullptr, 0,
        w_o16, xn, (long long)NQ_*D_, nullptr, 0, D_,
        nullptr, nullptr, 0, NQ_, D_, D_, 0);

    ln_kernel<<<dim3(B_*NQ_), blk256, 0, stream>>>(xn, queries, agg_g, agg_b,
                                                   xb, xb16);

    // --- Morton orders ----------------------------------------------------
    minmax_kernel<<<dim3(B_), blk256, 0, stream>>>(qpos, mnb, spanb);
    morton_kernel<<<dim3(NQ_/256, B_), blk256, 0, stream>>>(qpos, mnb, spanb, keys);
    sort_kernel<<<dim3(B_*2), dim3(1024), 0, stream>>>(keys, ordg, invb);

    // --- Mamba layers -----------------------------------------------------
    for (int l = 0; l < NL_; ++l) {
        ln_kernel<<<dim3(B_*NQ_), blk256, 0, stream>>>(xb, nullptr, pre_g, pre_b,
                                                       nullptr, xn16);

        gemm_f16wmma<<<dim3((PROJ_+63)/64, NQ_/64, B_*2), blk256, 0, stream>>>(
            xn16, D_, 0, ordg, (long long)NQ_,
            m_in_w16 + (size_t)l*PROJ_*D_, zx, (long long)NQ_*PROJ_,
            nullptr, 0, PROJ_,
            nullptr, nullptr, 0, NQ_, PROJ_, D_, 0);

        conv_kernel<<<dim3((NQ_*CONV_)/256, B_*2), blk256, 0, stream>>>(
            zx, m_conv_w + (size_t)l*CONV_*KER_, m_conv_b + (size_t)l*CONV_, xbc);

        dt_kernel<<<dim3((NQ_*NH_)/256, B_*2), blk256, 0, stream>>>(
            zx, m_dt_b + (size_t)l*NH_, dtbuf);

        scan_kernel<<<dim3(NH_, B_*2), blk256, 0, stream>>>(
            xbc, dtbuf, m_A_log + (size_t)l*NH_, m_Dp + (size_t)l*NH_, ysb);

        gated_rms_kernel<<<dim3(NQ_, B_*2), blk256, 0, stream>>>(
            ysb, zx, m_norm_w + (size_t)l*INTER_, hbuf16);

        gemm_f16wmma<<<dim3(D_/64, NQ_/64, B_*2), blk256, 0, stream>>>(
            hbuf16, INTER_, (long long)NQ_*INTER_, nullptr, 0,
            m_out_w16 + (size_t)l*D_*INTER_, ydir, (long long)NQ_*D_,
            nullptr, 0, D_,
            nullptr, nullptr, 0, NQ_, D_, INTER_, 0);

        combine_ln_kernel<<<dim3(NQ_, B_), blk256, 0, stream>>>(
            xb, xb16, ydir, invb, fin_g, fin_b);
    }

    // --- FFN + final LN ---------------------------------------------------
    gemm_f16wmma<<<dim3(HFF_/64, NQ_/64, B_), blk256, 0, stream>>>(
        xb16, D_, (long long)NQ_*D_, nullptr, 0,
        f_w116, nullptr, 0, ffh16, (long long)NQ_*HFF_, HFF_,
        f_b1, nullptr, 0, NQ_, HFF_, D_, 1);

    gemm_f16wmma<<<dim3(D_/64, NQ_/64, B_), blk256, 0, stream>>>(
        ffh16, HFF_, (long long)NQ_*HFF_, nullptr, 0,
        f_w216, ffo, (long long)NQ_*D_, nullptr, 0, D_,
        f_b2, xb, (long long)NQ_*D_, NQ_, D_, HFF_, 0);

    ln_kernel<<<dim3(B_*NQ_), blk256, 0, stream>>>(ffo, nullptr, fl_g, fl_b,
                                                   out, nullptr);
}
